// HLE_28853590295070
// MI455X (gfx1250) — matrix operand order repaired and checked by reading, not compiled, not run on hardware
//
#include <hip/hip_runtime.h>
#include <hip/hip_bf16.h>

// ---------------------------------------------------------------------------
// CDNA5 / gfx1250 implementation.
// Heavy math: bf16 WMMA GEMMs (V_WMMA_F32_16X16X32_BF16), fp32 accumulate.
// 128x256 block tile, 8 waves, 64x64 wave tile (4x4 WMMA frags) ->
// 16 WMMAs per K-step vs 16 ds_load_b128 (1.0 LDS loads / WMMA).
// BN statistics computed in fp32 on the pre-BN GEMM output.
// ---------------------------------------------------------------------------

typedef __bf16 bf16;
typedef __attribute__((ext_vector_type(16))) __bf16 bf16x16;
typedef __attribute__((ext_vector_type(8)))  float  floatx8;

#define TILE_M 128
#define TILE_N 256
#define TILE_K 32
#define LDS_K  40   // padded K stride (bf16): 80B rows -> conflict-free frag reads

union FragU {
    bf16x16 v;
    uint4   u[2];
};

// C[M,N] = A[M,K] (bf16, row-major) * Bt[N,K]^T (bf16, weights pre-transposed) + bias
__global__ __launch_bounds__(256)
void wmma_gemm_bias(const bf16* __restrict__ A,
                    const bf16* __restrict__ Bt,
                    const float* __restrict__ bias,
                    float* __restrict__ C,
                    int M, int N, int K)
{
    __shared__ alignas(16) bf16 As[2][TILE_M][LDS_K];   // 20 KB
    __shared__ alignas(16) bf16 Bs[2][TILE_N][LDS_K];   // 40 KB

    const int tid  = threadIdx.x;
    const int lane = tid & 31;
    const int wave = tid >> 5;
    const int wm   = wave >> 2;   // 0..1 : 64-row slice
    const int wn   = wave & 3;    // 0..3 : 64-col slice
    const int bm   = blockIdx.y * TILE_M;
    const int bn   = blockIdx.x * TILE_N;

    floatx8 acc[4][4];
#pragma unroll
    for (int i = 0; i < 4; ++i)
#pragma unroll
        for (int j = 0; j < 4; ++j)
            acc[i][j] = (floatx8){0.f,0.f,0.f,0.f,0.f,0.f,0.f,0.f};

    // global -> LDS stage.
    // A: 128 rows x 32 k ; thread -> row = tid>>1, 16 bf16 (two B128)
    // B: 256 rows x 32 k ; thread -> row = tid, 32 bf16 (four B128, contiguous)
    auto load_tiles = [&](int kbase, int buf) {
        {
            const int row  = tid >> 1;
            const int kofs = (tid & 1) * 16;
            const uint4* src = reinterpret_cast<const uint4*>(
                A + (size_t)(bm + row) * K + kbase + kofs);
            uint4* dst = reinterpret_cast<uint4*>(&As[buf][row][kofs]);
            dst[0] = src[0];
            dst[1] = src[1];
            if (kbase + 4 * TILE_K < K)
                __builtin_prefetch(A + (size_t)(bm + row) * K + kbase + 4 * TILE_K, 0, 1);
        }
        {
            const int row = tid;
            uint4 v0 = {0u,0u,0u,0u}, v1 = {0u,0u,0u,0u};
            uint4 v2 = {0u,0u,0u,0u}, v3 = {0u,0u,0u,0u};
            if (bn + row < N) {
                const uint4* src = reinterpret_cast<const uint4*>(
                    Bt + (size_t)(bn + row) * K + kbase);
                v0 = src[0]; v1 = src[1]; v2 = src[2]; v3 = src[3];
                if (kbase + 4 * TILE_K < K)
                    __builtin_prefetch(Bt + (size_t)(bn + row) * K + kbase + 4 * TILE_K, 0, 1);
            }
            uint4* dst = reinterpret_cast<uint4*>(&Bs[buf][row][0]);
            dst[0] = v0; dst[1] = v1; dst[2] = v2; dst[3] = v3;
        }
    };

    // A fragment (16x32 bf16): lane<16 -> K {0..7,16..23}; lane>=16 -> K {8..15,24..31}
    auto load_a_frag = [&](int buf, int mf) -> bf16x16 {
        FragU f;
        const int row = wm * 64 + mf * 16 + (lane & 15);
        const int klo = (lane >> 4) * 8;
        f.u[0] = *reinterpret_cast<const uint4*>(&As[buf][row][klo]);
        f.u[1] = *reinterpret_cast<const uint4*>(&As[buf][row][klo + 16]);
        return f.v;
    };
    // B fragment (32x16 bf16): lane = N col, lane<16 -> K {0..7,16..23}; lane>=16 -> K {8..15,24..31}
    auto load_b_frag = [&](int buf, int nf) -> bf16x16 {
        FragU f;
        const int row = wn * 64 + nf * 16 + (lane & 15);
        const int k0  = (lane >> 4) * 8;
        f.u[0] = *reinterpret_cast<const uint4*>(&Bs[buf][row][k0]);
        f.u[1] = *reinterpret_cast<const uint4*>(&Bs[buf][row][k0 + 16]);
        return f.v;
    };

    load_tiles(0, 0);
    __syncthreads();

    const int nk = K / TILE_K;
    for (int ks = 0; ks < nk; ++ks) {
        const int buf = ks & 1;
        if (ks + 1 < nk)
            load_tiles((ks + 1) * TILE_K, buf ^ 1);

        bf16x16 afrag[4], bfrag[4];
#pragma unroll
        for (int mf = 0; mf < 4; ++mf) afrag[mf] = load_a_frag(buf, mf);
#pragma unroll
        for (int nf = 0; nf < 4; ++nf) bfrag[nf] = load_b_frag(buf, nf);

#pragma unroll
        for (int mf = 0; mf < 4; ++mf)
#pragma unroll
            for (int nf = 0; nf < 4; ++nf)
                acc[mf][nf] = __builtin_amdgcn_wmma_f32_16x16x32_bf16(
                    false, afrag[mf], false, bfrag[nf],
                    (short)0, acc[mf][nf], false, false);

        __syncthreads();
    }

    // C/D layout: VGPR i holds M = i (lanes 0-15) or M = i+8 (lanes 16-31), N = lane&15
#pragma unroll
    for (int mf = 0; mf < 4; ++mf) {
#pragma unroll
        for (int nf = 0; nf < 4; ++nf) {
            const int rbase = bm + wm * 64 + mf * 16 + ((lane >> 4) << 3);
            const int c     = bn + wn * 64 + nf * 16 + (lane & 15);
            if (c < N) {
                const float bb = bias[c];
#pragma unroll
                for (int i = 0; i < 8; ++i)
                    C[(size_t)(rbase + i) * N + c] = acc[mf][nf][i] + bb;
            }
        }
    }
}

// Biased per-column mean/var over `rows` (BatchNorm1d training stats).
__global__ __launch_bounds__(256)
void colstats(const float* __restrict__ X, float* __restrict__ mean,
              float* __restrict__ var, int rows, int N)
{
    __shared__ float ssum[4][64];
    __shared__ float ssq[4][64];
    const int tx  = threadIdx.x & 63;
    const int ty  = threadIdx.x >> 6;
    const int col = blockIdx.x * 64 + tx;
    float s = 0.f, q = 0.f;
    if (col < N) {
        for (int r = ty; r < rows; r += 4) {
            const float v = X[(size_t)r * N + col];
            s += v;
            q += v * v;
        }
    }
    ssum[ty][tx] = s;
    ssq[ty][tx]  = q;
    __syncthreads();
    if (ty == 0 && col < N) {
        const float S = ssum[0][tx] + ssum[1][tx] + ssum[2][tx] + ssum[3][tx];
        const float Q = ssq[0][tx] + ssq[1][tx] + ssq[2][tx] + ssq[3][tx];
        const float m = S / (float)rows;
        mean[col] = m;
        var[col]  = Q / (float)rows - m * m;
    }
}

// y = (x - mean) * rsqrt(var+eps) * g + b ; optional ELU ; out fp32 or bf16.
template <bool ELU, bool OUT_BF16>
__global__ __launch_bounds__(256)
void bn_act(const float* __restrict__ X, const float* __restrict__ mean,
            const float* __restrict__ var, const float* __restrict__ gamma,
            const float* __restrict__ beta, void* __restrict__ out,
            int rows, int N)
{
    const size_t idx   = (size_t)blockIdx.x * blockDim.x + threadIdx.x;
    const size_t total = (size_t)rows * (size_t)N;
    if (idx >= total) return;
    const int c  = (int)(idx % (size_t)N);
    const float rs = rsqrtf(var[c] + 1e-5f);
    float y = (X[idx] - mean[c]) * rs * gamma[c] + beta[c];
    if (ELU) y = (y > 0.f) ? y : (__expf(y) - 1.f);
    if (OUT_BF16) ((bf16*)out)[idx] = (bf16)y;
    else          ((float*)out)[idx] = y;
}

__global__ __launch_bounds__(256)
void f32_to_bf16(const float* __restrict__ x, bf16* __restrict__ y, size_t n)
{
    const size_t i = (size_t)blockIdx.x * blockDim.x + threadIdx.x;
    if (i < n) y[i] = (bf16)x[i];
}

// W[K,N] fp32 -> Wt[N,K] bf16 (coalesced writes)
__global__ __launch_bounds__(256)
void f32_to_bf16_T(const float* __restrict__ W, bf16* __restrict__ Wt, int K, int N)
{
    const size_t i = (size_t)blockIdx.x * blockDim.x + threadIdx.x;
    if (i >= (size_t)K * (size_t)N) return;
    const int k = (int)(i % (size_t)K);
    const int n = (int)(i / (size_t)K);
    Wt[i] = (bf16)W[(size_t)k * N + n];
}

// final[b,leaf] = w0*o2[b,leaf] + w1*o0[b,leaf>>6] + w2*o1[b,leaf>>3] + bfin
// (exactly the M1/M2 indicator matmuls + stack/einsum of the reference)
__global__ __launch_bounds__(256)
void combine_final(const float* __restrict__ o0, const float* __restrict__ o1,
                   const float* __restrict__ o2, const float* __restrict__ wfin,
                   const float* __restrict__ bfin, float* __restrict__ Yf, int rows)
{
    const size_t i     = (size_t)blockIdx.x * blockDim.x + threadIdx.x;
    const size_t total = (size_t)rows * 512;
    if (i >= total) return;
    const int    leaf = (int)(i & 511);
    const size_t b    = i >> 9;
    const float v = o2[i] * wfin[0]
                  + o0[b * 8  + (leaf >> 6)] * wfin[1]
                  + o1[b * 64 + (leaf >> 3)] * wfin[2];
    Yf[i] = v + bfin[0];
}

// ---------------------------------------------------------------------------
// Host side
// ---------------------------------------------------------------------------
extern "C" void kernel_launch(void* const* d_in, const int* in_sizes, int n_in,
                              void* d_out, int out_size, void* d_ws, size_t ws_size,
                              hipStream_t stream)
{
    (void)in_sizes; (void)n_in; (void)out_size; (void)ws_size;

    const int Mrows = 8192, NIN = 2048, NHID = 2048;
    const int NOUT[3] = {8, 64, 512};
    // d_out: [outs0 | outs1 | outs2 | final], flat fp32
    const size_t outOff[4] = {
        0,
        (size_t)Mrows * 8,
        (size_t)Mrows * 8 + (size_t)Mrows * 64,
        (size_t)Mrows * 8 + (size_t)Mrows * 64 + (size_t)Mrows * 512
    };

    // Inputs in setup_inputs() insertion order:
    // 0: inputs; then per level l (base 1+12l): W1,b1,g1,be1,W2,b2,g2,be2,W3,b3,g3,be3;
    // 37: w_fin, 38: b_fin, 39: g_fin, 40: be_fin, 41: M1, 42: M2 (M1/M2 replaced by gathers)
    const float* X     = (const float*)d_in[0];
    const float* wfin  = (const float*)d_in[37];
    const float* bfin  = (const float*)d_in[38];
    const float* gfin  = (const float*)d_in[39];
    const float* befin = (const float*)d_in[40];

    char*  ws  = (char*)d_ws;
    size_t off = 0;
    auto carve = [&](size_t bytes) -> void* {
        void* p = ws + off;
        off += (bytes + 255) & ~(size_t)255;
        return p;
    };
    bf16*  Xbf  = (bf16*)carve((size_t)Mrows * NIN * 2);   //  32 MB
    bf16*  Wt1  = (bf16*)carve((size_t)NHID * NIN * 2);    //   8 MB
    bf16*  Wt2  = (bf16*)carve((size_t)NHID * NHID * 2);   //   8 MB
    bf16*  Wt3  = (bf16*)carve((size_t)512 * NHID * 2);    //   2 MB
    float* Y    = (float*)carve((size_t)Mrows * NHID * 4); //  64 MB (reused every layer)
    bf16*  Hbf  = (bf16*)carve((size_t)Mrows * NHID * 2);  //  32 MB (reused)
    float* mean = (float*)carve((size_t)NHID * 4);
    float* var  = (float*)carve((size_t)NHID * 4);

    {   // convert input once
        const size_t n = (size_t)Mrows * NIN;
        f32_to_bf16<<<(n + 255) / 256, 256, 0, stream>>>(X, Xbf, n);
    }

    float* outF = (float*)d_out;
    for (int l = 0; l < 3; ++l) {
        const int ib = 1 + 12 * l;
        const float* W1  = (const float*)d_in[ib + 0];
        const float* b1  = (const float*)d_in[ib + 1];
        const float* g1  = (const float*)d_in[ib + 2];
        const float* be1 = (const float*)d_in[ib + 3];
        const float* W2  = (const float*)d_in[ib + 4];
        const float* b2  = (const float*)d_in[ib + 5];
        const float* g2  = (const float*)d_in[ib + 6];
        const float* be2 = (const float*)d_in[ib + 7];
        const float* W3  = (const float*)d_in[ib + 8];
        const float* b3  = (const float*)d_in[ib + 9];
        const float* g3  = (const float*)d_in[ib + 10];
        const float* be3 = (const float*)d_in[ib + 11];
        const int N3 = NOUT[l];

        { const size_t n = (size_t)NIN * NHID;
          f32_to_bf16_T<<<(n + 255) / 256, 256, 0, stream>>>(W1, Wt1, NIN, NHID); }
        { const size_t n = (size_t)NHID * NHID;
          f32_to_bf16_T<<<(n + 255) / 256, 256, 0, stream>>>(W2, Wt2, NHID, NHID); }
        { const size_t n = (size_t)NHID * N3;
          f32_to_bf16_T<<<(n + 255) / 256, 256, 0, stream>>>(W3, Wt3, NHID, N3); }

        // layer 1: X @ W1 -> Y ; BN+ELU -> Hbf
        wmma_gemm_bias<<<dim3((NHID + TILE_N - 1) / TILE_N, Mrows / TILE_M), 256, 0, stream>>>(
            Xbf, Wt1, b1, Y, Mrows, NHID, NIN);
        colstats<<<NHID / 64, 256, 0, stream>>>(Y, mean, var, Mrows, NHID);
        { const size_t n = (size_t)Mrows * NHID;
          bn_act<true, true><<<(n + 255) / 256, 256, 0, stream>>>(
              Y, mean, var, g1, be1, Hbf, Mrows, NHID); }

        // layer 2: H1 @ W2 -> Y ; BN+ELU -> Hbf
        wmma_gemm_bias<<<dim3((NHID + TILE_N - 1) / TILE_N, Mrows / TILE_M), 256, 0, stream>>>(
            Hbf, Wt2, b2, Y, Mrows, NHID, NHID);
        colstats<<<NHID / 64, 256, 0, stream>>>(Y, mean, var, Mrows, NHID);
        { const size_t n = (size_t)Mrows * NHID;
          bn_act<true, true><<<(n + 255) / 256, 256, 0, stream>>>(
              Y, mean, var, g2, be2, Hbf, Mrows, NHID); }

        // head: H2 @ W3 -> Y[:, :N3] ; BN -> d_out level slot
        wmma_gemm_bias<<<dim3((N3 + TILE_N - 1) / TILE_N, Mrows / TILE_M), 256, 0, stream>>>(
            Hbf, Wt3, b3, Y, Mrows, N3, NHID);
        colstats<<<(N3 + 63) / 64, 256, 0, stream>>>(Y, mean, var, Mrows, N3);
        { const size_t n = (size_t)Mrows * N3;
          bn_act<false, false><<<(n + 255) / 256, 256, 0, stream>>>(
              Y, mean, var, g3, be3, outF + outOff[l], Mrows, N3); }
    }

    // final combine + BN
    {
        const size_t n = (size_t)Mrows * 512;
        combine_final<<<(n + 255) / 256, 256, 0, stream>>>(
            outF + outOff[0], outF + outOff[1], outF + outOff[2], wfin, bfin, Y, Mrows);
        colstats<<<512 / 64, 256, 0, stream>>>(Y, mean, var, Mrows, 512);
        bn_act<false, false><<<(n + 255) / 256, 256, 0, stream>>>(
            Y, mean, var, gfin, befin, outF + outOff[3], Mrows, 512);
    }
}
